// LSTMEncoder_63453846831722
// MI455X (gfx1250) — matrix operand order repaired and checked by reading, not compiled, not run on hardware
//
#include <hip/hip_runtime.h>

// ---------------------------------------------------------------------------
// CDNA5 (gfx1250) bidirectional 2-layer LSTM encoder, bf16 WMMA everywhere.
//   B=512, T=512, I=16, H=256, L=64.  wave32, 16x16x32 bf16 WMMA, f32 accum.
// Phases:
//   1) cvt_wih1      : layer-1 input weights f32 -> bf16 (L2-resident, 2 MB)
//   2) lstm_rec<0>   : layer-0 recurrence, w_hh resident in VGPRs as WMMA-B
//                      fragments, x-GEMM (K=16, zero-padded to 32) fused,
//                      writes x1 = concat(out0f,out0b) bf16 [B*T,512]
//   3) gx_gemm       : gx1 = x1 @ w_ih1^T + b (both dirs), A async-staged LDS
//   4) lstm_rec<1>   : layer-1 recurrence from gx1 (pipelined gx loads),
//                      writes final hiddens
//   5) fc_kern       : out = concat(h1f,h1b) @ fc_w^T + fc_b
// ---------------------------------------------------------------------------

typedef __attribute__((ext_vector_type(16))) __bf16 v16bf;
typedef __attribute__((ext_vector_type(8)))  float  v8f;

union Frag {
  v16bf  v;
  __bf16 e[16];
  uint4  q[2];
};

__device__ __forceinline__ v8f wmma_bf16(const Frag& a, const Frag& b, v8f c) {
  return __builtin_amdgcn_wmma_f32_16x16x32_bf16(false, a.v, false, b.v,
                                                 (short)0, c, false, false);
}

__device__ __forceinline__ float fsig(float x)  { return 1.0f / (1.0f + __expf(-x)); }
__device__ __forceinline__ float ftanh(float x) { return 2.0f * fsig(2.0f * x) - 1.0f; }

// CDNA5 async global->LDS copy (ASYNCcnt-tracked, no VGPR round trip).
__device__ __forceinline__ void async_ld_b128(void* lds_ptr, const void* gptr) {
  const unsigned int       loff = (unsigned int)(unsigned long long)lds_ptr;
  const unsigned long long ga   = (unsigned long long)gptr;
  asm volatile("global_load_async_to_lds_b128 %0, %1, off"
               :: "v"(loff), "v"(ga) : "memory");
}
__device__ __forceinline__ void wait_async0() {
  asm volatile("s_wait_asynccnt 0x0" ::: "memory");
}

// ---------------------------------------------------------------------------
// Recurrent kernel. MODE 0: layer 0 (fused x-GEMM, writes whole sequence).
//                   MODE 1: layer 1 (reads precomputed gx, writes final h).
// grid = (32 batch-tiles, 2 directions), block = 512 (16 waves).
// Wave w owns hidden units j in [16w, 16w+16); holds w_hh B-fragments in VGPRs.
// ---------------------------------------------------------------------------
template <int MODE>
__global__ __launch_bounds__(512) void lstm_rec(
    const float* __restrict__ x,
    const float* __restrict__ wih_f, const float* __restrict__ wih_b,
    const __bf16* __restrict__ gx_f, const __bf16* __restrict__ gx_b,
    const float* __restrict__ whh_f, const float* __restrict__ whh_b,
    const float* __restrict__ bias_f, const float* __restrict__ bias_b,
    __bf16* __restrict__ x1out, float* __restrict__ hfin)
{
  constexpr int T = 512, B = 512, H = 256;
  constexpr int HP = H + 8;                 // padded LDS stride: 4-bank row skew
  __shared__ __bf16 hbuf[2][16 * HP];       // double-buffered h tile (bf16)

  const int tid  = threadIdx.x;
  const int lane = tid & 31;
  const int wv   = tid >> 5;                // 0..15
  const int ln   = lane & 15;
  const int lh   = lane >> 4;               // lane half selects K group
  const int dir  = blockIdx.y;              // 0 = forward, 1 = backward
  const int bb   = blockIdx.x * 16;         // batch tile base
  const int jb   = wv * 16;                 // hidden-unit tile base

  const float*  whh  = dir ? whh_b  : whh_f;
  const float*  wih  = dir ? wih_b  : wih_f;
  const float*  bias = dir ? bias_b : bias_f;
  const __bf16* gx   = dir ? gx_b   : gx_f;

  // Resident recurrent weights: B-frag layout = w_hh[n][k] with n = column
  // (lane&15), K-half = lane>>4, 2 packed bf16 K values per VGPR.
  Frag wb[4][8];
  #pragma unroll
  for (int g = 0; g < 4; ++g)
    #pragma unroll
    for (int kc = 0; kc < 8; ++kc) {
      const float* p = whh + (size_t)(g * H + jb + ln) * H + kc * 32 + lh * 16;
      #pragma unroll
      for (int i = 0; i < 16; ++i) wb[g][kc].e[i] = (__bf16)p[i];
    }

  // Layer-0 input weights: K=16 real, K=16..31 zero padded.
  Frag  wib[4];
  float bs[4];
  #pragma unroll
  for (int g = 0; g < 4; ++g) {
    bs[g] = bias ? bias[g * H + jb + ln] : 0.0f;
    #pragma unroll
    for (int i = 0; i < 16; ++i) wib[g].e[i] = (__bf16)0.0f;
    if (MODE == 0) {
      const float* p = wih + (size_t)(g * H + jb + ln) * 16 + lh * 8;
      #pragma unroll
      for (int i = 0; i < 8; ++i) wib[g].e[i] = (__bf16)p[i];
    }
  }

  // Software-pipelined gx loads (MODE 1): fetch step s+1 while step s computes.
  float gpre[4][8];
  auto load_gx = [&](int t) {
    #pragma unroll
    for (int g = 0; g < 4; ++g)
      #pragma unroll
      for (int v = 0; v < 8; ++v) {
        const size_t row = (size_t)(bb + lh * 8 + v) * T + t;
        gpre[g][v] = (float)gx[row * 1024 + g * H + jb + ln];
      }
  };
  if (MODE == 1) load_gx(dir ? T - 1 : 0);

  v8f cc = {};                              // cell-state tile, C layout
  for (int i = tid; i < 16 * HP; i += 512) hbuf[0][i] = (__bf16)0.0f;
  __syncthreads();

  for (int s = 0; s < T; ++s) {
    const int t    = dir ? (T - 1 - s) : s;
    const int rb   = s & 1;
    const int wbuf = rb ^ 1;

    v8f acc[4];
    #pragma unroll
    for (int g = 0; g < 4; ++g)
      #pragma unroll
      for (int v = 0; v < 8; ++v)
        acc[g][v] = (MODE == 1) ? (bs[g] + gpre[g][v]) : bs[g];

    if (MODE == 0) {                        // fused x-GEMM, K=16 padded to 32
      Frag xa;
      #pragma unroll
      for (int i = 8; i < 16; ++i) xa.e[i] = (__bf16)0.0f;
      const float* xp = x + ((size_t)(bb + ln) * T + t) * 16 + lh * 8;
      #pragma unroll
      for (int i = 0; i < 8; ++i) xa.e[i] = (__bf16)xp[i];
      #pragma unroll
      for (int g = 0; g < 4; ++g) acc[g] = wmma_bf16(xa, wib[g], acc[g]);
    }

    // h @ w_hh^T : A fragments broadcast from LDS, 8 K-chunks x 4 gates
    #pragma unroll
    for (int kc = 0; kc < 8; ++kc) {
      Frag ha;
      ha.q[0] = *(const uint4*)&hbuf[rb][ln * HP + kc * 32 + lh * 16];
      ha.q[1] = *(const uint4*)&hbuf[rb][ln * HP + kc * 32 + lh * 16 + 8];
      #pragma unroll
      for (int g = 0; g < 4; ++g) acc[g] = wmma_bf16(ha, wb[g][kc], acc[g]);
    }

    // Prefetch next step's gates while WMMA results drain / barrier waits.
    if (MODE == 1 && s + 1 < T) load_gx(dir ? (T - 2 - s) : (s + 1));

    // gate nonlinearity + state update (i,f,g,o torch order)
    #pragma unroll
    for (int v = 0; v < 8; ++v) {
      const float iv = fsig(acc[0][v]);
      const float fv = fsig(acc[1][v]);
      const float gv = ftanh(acc[2][v]);
      const float ov = fsig(acc[3][v]);
      const float cv = fv * cc[v] + iv * gv;
      cc[v] = cv;
      const float hv = ov * ftanh(cv);
      const int row = lh * 8 + v;
      hbuf[wbuf][row * HP + jb + ln] = (__bf16)hv;
      if (MODE == 0)
        x1out[((size_t)(bb + row) * T + t) * 512 + dir * 256 + jb + ln] = (__bf16)hv;
      if (MODE == 1 && s == T - 1)
        hfin[((size_t)dir * B + bb + row) * H + jb + ln] = hv;
    }
    __syncthreads();
  }
}

// ---------------------------------------------------------------------------
// gx1 = x1 @ w_ih1^T + b for both directions.
// grid = B*T/256, block = 512 (16 waves). A tile [256 x 512] bf16 async-staged
// into dynamic LDS (260 KB, padded stride). Weights streamed from L2 (2 MB).
// ---------------------------------------------------------------------------
__global__ __launch_bounds__(512) void gx_gemm(
    const __bf16* __restrict__ x1,
    const __bf16* __restrict__ w_f, const __bf16* __restrict__ w_b,
    const float* __restrict__ bias_f, const float* __restrict__ bias_b,
    __bf16* __restrict__ gx_out_f, __bf16* __restrict__ gx_out_b)
{
  constexpr int MB = 256;                   // (b,t) rows per workgroup
  constexpr int KP = 512 + 8;               // padded LDS stride
  extern __shared__ __bf16 atile[];

  const int tid  = threadIdx.x;
  const int lane = tid & 31;
  const int wv   = tid >> 5;
  const int ln   = lane & 15;
  const int lh   = lane >> 4;
  const size_t rowbase = (size_t)blockIdx.x * MB;

  // Async-stage A tile (256 KB) into padded LDS: ASYNCcnt path, no VGPR bounce.
  const uint4* src = (const uint4*)(x1 + rowbase * 512);
  for (int idx = tid; idx < MB * 512 / 8; idx += 512) {
    const int e = idx * 8;
    const int r = e >> 9;
    const int c = e & 511;
    async_ld_b128(&atile[r * KP + c], src + idx);
  }
  wait_async0();
  __syncthreads();

  const int nt = wv & 3;                    // n-tile within 64-wide n-block
  const int mg = wv >> 2;                   // owns 4 m-tiles

  for (int nb = 0; nb < 32; ++nb) {         // 16 n-blocks x 2 directions
    const int d = nb >> 4;
    const __bf16* w    = d ? w_b      : w_f;
    const float*  bias = d ? bias_b   : bias_f;
    __bf16*       gxo  = d ? gx_out_b : gx_out_f;
    const int   ncol = (nb & 15) * 64 + nt * 16;
    const float bv   = bias[ncol + ln];

    v8f acc[4];
    #pragma unroll
    for (int i = 0; i < 4; ++i)
      #pragma unroll
      for (int v = 0; v < 8; ++v) acc[i][v] = bv;

    for (int kc = 0; kc < 16; ++kc) {
      Frag bf;
      const __bf16* wp = w + (size_t)(ncol + ln) * 512 + kc * 32 + lh * 16;
      bf.q[0] = *(const uint4*)wp;
      bf.q[1] = *(const uint4*)(wp + 8);
      #pragma unroll
      for (int i = 0; i < 4; ++i) {
        Frag a;
        const int r0 = (mg * 4 + i) * 16 + ln;
        a.q[0] = *(const uint4*)&atile[r0 * KP + kc * 32 + lh * 16];
        a.q[1] = *(const uint4*)&atile[r0 * KP + kc * 32 + lh * 16 + 8];
        acc[i] = wmma_bf16(a, bf, acc[i]);
      }
    }

    #pragma unroll
    for (int i = 0; i < 4; ++i)
      #pragma unroll
      for (int v = 0; v < 8; ++v) {
        const int row = (mg * 4 + i) * 16 + lh * 8 + v;
        gxo[(rowbase + row) * 1024 + ncol + ln] = (__bf16)acc[i][v];
      }
  }
}

// ---------------------------------------------------------------------------
__global__ void cvt_wih1(const float* __restrict__ a, const float* __restrict__ b,
                         __bf16* __restrict__ oa, __bf16* __restrict__ ob)
{
  const int i = blockIdx.x * blockDim.x + threadIdx.x;   // 1024*512 exact
  oa[i] = (__bf16)a[i];
  ob[i] = (__bf16)b[i];
}

__global__ void fc_kern(const float* __restrict__ hfin, const float* __restrict__ w,
                        const float* __restrict__ bias, float* __restrict__ out)
{
  const int idx = blockIdx.x * blockDim.x + threadIdx.x;  // B*64
  const int b = idx >> 6, l = idx & 63;
  const float* hf = hfin + (size_t)b * 256;
  const float* hb = hfin + 512ull * 256ull + (size_t)b * 256;
  float acc = bias[l];
  #pragma unroll 4
  for (int k = 0; k < 256; ++k)
    acc += hf[k] * w[l * 512 + k] + hb[k] * w[l * 512 + 256 + k];
  out[idx] = acc;
}

// ---------------------------------------------------------------------------
extern "C" void kernel_launch(void* const* d_in, const int* in_sizes, int n_in,
                              void* d_out, int out_size, void* d_ws, size_t ws_size,
                              hipStream_t stream)
{
  const float* x     = (const float*)d_in[0];
  const float* wih0f = (const float*)d_in[1];
  const float* whh0f = (const float*)d_in[2];
  const float* b0f   = (const float*)d_in[3];
  const float* wih0b = (const float*)d_in[4];
  const float* whh0b = (const float*)d_in[5];
  const float* b0b   = (const float*)d_in[6];
  const float* wih1f = (const float*)d_in[7];
  const float* whh1f = (const float*)d_in[8];
  const float* b1f   = (const float*)d_in[9];
  const float* wih1b = (const float*)d_in[10];
  const float* whh1b = (const float*)d_in[11];
  const float* b1b   = (const float*)d_in[12];
  const float* fcw   = (const float*)d_in[13];
  const float* fcb   = (const float*)d_in[14];

  // Workspace layout (~1.29 GB): x1 | gx1f | gx1b | wih1 bf16 | final hiddens
  constexpr size_t BT = 512ull * 512ull;
  __bf16* x1   = (__bf16*)d_ws;
  __bf16* gx1f = x1 + BT * 512;
  __bf16* gx1b = gx1f + BT * 1024;
  __bf16* w1fb = gx1b + BT * 1024;
  __bf16* w1bb = w1fb + 1024 * 512;
  float*  hfin = (float*)(w1bb + 1024 * 512);

  cvt_wih1<<<2048, 256, 0, stream>>>(wih1f, wih1b, w1fb, w1bb);

  lstm_rec<0><<<dim3(32, 2), 512, 0, stream>>>(
      x, wih0f, wih0b, nullptr, nullptr, whh0f, whh0b, b0f, b0b, x1, nullptr);

  constexpr int GX_LDS = (256 * (512 + 8)) * 2;   // 260 KB dynamic LDS
  hipFuncSetAttribute(reinterpret_cast<const void*>(gx_gemm),
                      hipFuncAttributeMaxDynamicSharedMemorySize, GX_LDS);
  gx_gemm<<<BT / 256, 512, GX_LDS, stream>>>(x1, w1fb, w1bb, b1f, b1b, gx1f, gx1b);

  lstm_rec<1><<<dim3(32, 2), 512, 0, stream>>>(
      nullptr, nullptr, nullptr, gx1f, gx1b, whh1f, whh1b, nullptr, nullptr,
      nullptr, hfin);

  fc_kern<<<(512 * 64) / 256, 256, 0, stream>>>(hfin, fcw, fcb, (float*)d_out);
}
